// BAC_15152644620305
// MI455X (gfx1250) — compile-verified, matrix operand order checked by reading, not run on hardware
//
#include <hip/hip_runtime.h>
#include <math.h>

// Problem constants (match reference)
#define LSEQ   2048
#define BATCH  8
#define D_IN   600
#define D_PAD  640       // 20*32: multiple of 64 (N-tile) and 32 (K-step)
#define U_IN   300
#define U_PAD  320       // 10*32
#define KFM    5
#define BL     (BATCH*LSEQ)

#define GEMM_THREADS 128  // 4 wave32s

typedef __attribute__((ext_vector_type(16))) __bf16 v16bf;
typedef __attribute__((ext_vector_type(8)))  float  v8f;

static __device__ __forceinline__ unsigned short f2bf(float f) {
  unsigned int u = __float_as_uint(f);
  u += 0x7FFFu + ((u >> 16) & 1u);           // round-to-nearest-even
  return (unsigned short)(u >> 16);
}

union FragCvt { unsigned int u[8]; v16bf v; };

// LDS tiles are stored "fragment-ready": for each 16-row block of a (rows x 32
// bf16) tile, lane l's 8 fragment dwords are contiguous (lane stride 32B).
// Per ISA 7.12.2 (16-bit A/B 16x32): lane l holds row (l&15); lane-half
// h=l>>4 covers dword cols {4h..4h+3} and {8+4h..8+4h+3}. So global b128
// chunk (row r, seg s: dword cols 4s..4s+3) belongs to lane (r&15)+((s&1)<<4),
// dword group s>>1. One 16-row block = 32 lanes * 8 dwords = 256 dwords.
static __device__ __forceinline__ v16bf load_frag_blk(const unsigned int* blk,
                                                      int lane) {
  const uint4* p = reinterpret_cast<const uint4*>(blk + lane * 8);
  uint4 x0 = p[0];
  uint4 x1 = p[1];
  FragCvt f;
  f.u[0] = x0.x; f.u[1] = x0.y; f.u[2] = x0.z; f.u[3] = x0.w;
  f.u[4] = x1.x; f.u[5] = x1.y; f.u[6] = x1.z; f.u[7] = x1.w;
  return f.v;
}

// C[M,N] = epilogue( A[M,K] @ Brows[N,K]^T )
// MODE 0: +bias[n], relu, store bf16   (dense layers)
// MODE 1: *scale, store f32            (affinity)
// MODE 2: store f32                    (alignment GEMMs)
// Block tile 128x64, 4 waves of 64x32 (4x2 WMMA 16x16x32 bf16 tiles).
// Double-buffered GLOBAL_LOAD_ASYNC_TO_LDS_B128 staging (ASYNCcnt): the
// next K-tile's copies are in flight while the current tile's WMMAs run.
template <int MODE>
__global__ __launch_bounds__(GEMM_THREADS) void gemm_wmma(
    const unsigned short* __restrict__ A, const unsigned short* __restrict__ B,
    void* __restrict__ Cvoid, const float* __restrict__ bias, float scale,
    int K, int lda, int ldb, int ldc,
    long long strideA, long long strideB, long long strideC) {
  __shared__ unsigned int As[2][8 * 256];  // 2 x (128 rows x 32 bf16)
  __shared__ unsigned int Bs[2][4 * 256];  // 2 x (64 rows x 32 bf16)

  const int tid  = threadIdx.x;
  const int lane = tid & 31;
  const int wave = tid >> 5;            // 0..3
  const int wm = (wave & 1) * 64;       // wave row offset in block tile
  const int wn = (wave >> 1) * 32;      // wave col offset in block tile
  const int m0 = blockIdx.y * 128;
  const int n0 = blockIdx.x * 64;
  const long long zb = blockIdx.z;
  const unsigned short* Ab = A + zb * strideA + (long long)m0 * lda;
  const unsigned short* Bb = B + zb * strideB + (long long)n0 * ldb;

  // Per-thread async-copy slots: LDS byte address (buffer 0) + global offset.
  const unsigned int aBufStride =
      (unsigned int)((const char*)&As[1][0] - (const char*)&As[0][0]);
  const unsigned int bBufStride =
      (unsigned int)((const char*)&Bs[1][0] - (const char*)&Bs[0][0]);
  unsigned int ldsA[4], voffA[4];
#pragma unroll
  for (int it = 0; it < 4; it++) {
    int idx = tid + it * GEMM_THREADS;   // 512 b128 chunks: 128 rows x 4 segs
    int r = idx >> 2, seg = idx & 3;
    int lp = (r & 15) + ((seg & 1) << 4);
    int dg = seg >> 1;
    ldsA[it] = (unsigned int)(unsigned long long)(const void*)
                   (&As[0][0] + (r >> 4) * 256 + lp * 8 + dg * 4);
    voffA[it] = (unsigned int)((r * lda + seg * 8) * 2);
  }
  unsigned int ldsB[2], voffB[2];
#pragma unroll
  for (int it = 0; it < 2; it++) {
    int idx = tid + it * GEMM_THREADS;   // 256 b128 chunks: 64 rows x 4 segs
    int r = idx >> 2, seg = idx & 3;
    int lp = (r & 15) + ((seg & 1) << 4);
    int dg = seg >> 1;
    ldsB[it] = (unsigned int)(unsigned long long)(const void*)
                   (&Bs[0][0] + (r >> 4) * 256 + lp * 8 + dg * 4);
    voffB[it] = (unsigned int)((r * ldb + seg * 8) * 2);
  }

  auto issue_tile = [&](int buf, const unsigned short* aK,
                        const unsigned short* bK) {
    unsigned int ao = buf ? aBufStride : 0u;
    unsigned int bo = buf ? bBufStride : 0u;
#pragma unroll
    for (int it = 0; it < 4; it++)
      asm volatile("global_load_async_to_lds_b128 %0, %1, %2"
                   :: "v"(ldsA[it] + ao), "v"(voffA[it]), "s"(aK) : "memory");
#pragma unroll
    for (int it = 0; it < 2; it++)
      asm volatile("global_load_async_to_lds_b128 %0, %1, %2"
                   :: "v"(ldsB[it] + bo), "v"(voffB[it]), "s"(bK) : "memory");
  };

  v8f acc[4][2];
#pragma unroll
  for (int i = 0; i < 4; i++)
#pragma unroll
    for (int j = 0; j < 2; j++)
#pragma unroll
      for (int e = 0; e < 8; e++) acc[i][j][e] = 0.f;

  issue_tile(0, Ab, Bb);
  int buf = 0;
  for (int k0 = 0; k0 < K; k0 += 32) {
    // Drain the copies targeting buf, make them visible block-wide.
    asm volatile("s_wait_asynccnt 0x0" ::: "memory");
    __syncthreads();
    // Prefetch the next K-tile into the other buffer while we compute.
    int nk = k0 + 32;
    if (nk < K) issue_tile(buf ^ 1, Ab + nk, Bb + nk);

    const unsigned int* aBase = &As[buf][0];
    const unsigned int* bBase = &Bs[buf][0];
    v16bf af[4], bfr[2];
#pragma unroll
    for (int mi = 0; mi < 4; mi++)
      af[mi] = load_frag_blk(aBase + ((wm >> 4) + mi) * 256, lane);
#pragma unroll
    for (int ni = 0; ni < 2; ni++)
      bfr[ni] = load_frag_blk(bBase + ((wn >> 4) + ni) * 256, lane);

#pragma unroll
    for (int mi = 0; mi < 4; mi++)
#pragma unroll
      for (int ni = 0; ni < 2; ni++)
        acc[mi][ni] = __builtin_amdgcn_wmma_f32_16x16x32_bf16(
            false, af[mi], false, bfr[ni], (short)0, acc[mi][ni], false, false);
    buf ^= 1;
  }

  // Epilogue. C 16x16 f32 layout: lane holds col N=(lane&15); lane-half adds 8
  // to the row; VGPR v = row offset 0..7.
  const int col_l = lane & 15;
  const int row_l = (lane >> 4) * 8;
#pragma unroll
  for (int mi = 0; mi < 4; mi++)
#pragma unroll
    for (int ni = 0; ni < 2; ni++) {
      int n = n0 + wn + ni * 16 + col_l;
      int mbase = m0 + wm + mi * 16 + row_l;
#pragma unroll
      for (int v = 0; v < 8; v++) {
        float val = acc[mi][ni][v];
        long long off = strideC * zb + (long long)(mbase + v) * ldc + n;
        if (MODE == 0) {
          val += bias[n];
          val = fmaxf(val, 0.f);
          ((unsigned short*)Cvoid)[off] = f2bf(val);
        } else if (MODE == 1) {
          ((float*)Cvoid)[off] = val * scale;
        } else {
          ((float*)Cvoid)[off] = val;
        }
      }
    }
}

// Convert stack_input f32 -> bf16, padded [L][640], plus transposed [640][L]
// via an LDS tile transpose. z: src = z&1, b = z>>1.
__global__ __launch_bounds__(256) void convert_inputs(
    const float* __restrict__ stack, unsigned short* __restrict__ Xbf,
    unsigned short* __restrict__ XTbf) {
  __shared__ unsigned short tile[64][65];
  const int z = blockIdx.z;
  const int src = z & 1, b = z >> 1;
  const long long img = (long long)src * BATCH + b;
  const float* base = stack + img * LSEQ * D_IN;
  unsigned short* xb = Xbf + img * (long long)LSEQ * D_PAD;
  unsigned short* xt = XTbf + img * (long long)D_PAD * LSEQ;
  const int l0 = blockIdx.x * 64, d0 = blockIdx.y * 64;
  const int tx = threadIdx.x & 63, ty = threadIdx.x >> 6;

#pragma unroll
  for (int i = 0; i < 16; i++) {
    int lr = ty + i * 4;
    int l = l0 + lr, d = d0 + tx;
    float v = (d < D_IN) ? base[(long long)l * D_IN + d] : 0.f;
    unsigned short bv = f2bf(v);
    xb[(long long)l * D_PAD + d] = bv;
    tile[tx][lr] = bv;  // tile[d_local][l_local]
  }
  __syncthreads();
#pragma unroll
  for (int i = 0; i < 16; i++) {
    int dr = ty + i * 4;
    xt[(long long)(d0 + dr) * LSEQ + l0 + tx] = tile[dr][tx];
  }
}

// W1/W2 [600,300] -> transposed padded bf16 [2][320][640]
__global__ void convert_weights(const float* __restrict__ W1,
                                const float* __restrict__ W2,
                                unsigned short* __restrict__ Wt) {
  int idx = blockIdx.x * blockDim.x + threadIdx.x;
  if (idx >= 2 * U_PAD * D_PAD) return;
  int which = idx / (U_PAD * D_PAD);
  int rem = idx % (U_PAD * D_PAD);
  int u = rem / D_PAD, d = rem % D_PAD;
  const float* W = which ? W2 : W1;
  float v = (u < U_IN && d < D_IN) ? W[(long long)d * U_IN + u] : 0.f;
  Wt[idx] = f2bf(v);
}

__global__ void convert_bias(const float* __restrict__ b1,
                             const float* __restrict__ b2,
                             float* __restrict__ bp) {
  int t = blockIdx.x * blockDim.x + threadIdx.x;
  if (t < U_PAD) bp[t] = (t < U_IN) ? b1[t] : 0.f;
  else if (t < 2 * U_PAD) { int u = t - U_PAD; bp[t] = (u < U_IN) ? b2[u] : 0.f; }
}

// Per-row max & sum(exp) of affinity (softmax over q). One block per row.
__global__ __launch_bounds__(256) void row_stats(const float* __restrict__ aff,
                                                 float* __restrict__ rmax,
                                                 float* __restrict__ rsum) {
  __shared__ float sm[8];
  __shared__ float bc;
  const int p = blockIdx.x, b = blockIdx.y;
  const float* row = aff + ((long long)b * LSEQ + p) * LSEQ;
  const int t = threadIdx.x, lane = t & 31, wave = t >> 5;
  float m = -3.4e38f;
  for (int q = t; q < LSEQ; q += 256) m = fmaxf(m, row[q]);
  for (int o = 16; o; o >>= 1) m = fmaxf(m, __shfl_xor(m, o, 32));
  if (lane == 0) sm[wave] = m;
  __syncthreads();
  if (t == 0) {
    float mm = sm[0];
    for (int w = 1; w < 8; w++) mm = fmaxf(mm, sm[w]);
    bc = mm;
  }
  __syncthreads();
  m = bc;
  float s = 0.f;
  for (int q = t; q < LSEQ; q += 256) s += __expf(row[q] - m);
  for (int o = 16; o; o >>= 1) s += __shfl_xor(s, o, 32);
  if (lane == 0) sm[wave] = s;
  __syncthreads();
  if (t == 0) {
    float ss = 0.f;
    for (int w = 0; w < 8; w++) ss += sm[w];
    rmax[(long long)b * LSEQ + p] = m;
    rsum[(long long)b * LSEQ + p] = ss;
  }
}

// Per-column max & sum(exp) (softmax over p). Coalesced column streaming.
__global__ __launch_bounds__(256) void col_stats(const float* __restrict__ aff,
                                                 float* __restrict__ cmax,
                                                 float* __restrict__ csum) {
  const int q = blockIdx.x * 256 + threadIdx.x;
  const int b = blockIdx.y;
  const float* base = aff + (long long)b * LSEQ * LSEQ + q;
  float m = -3.4e38f;
  for (int p = 0; p < LSEQ; p++) m = fmaxf(m, base[(long long)p * LSEQ]);
  float s = 0.f;
  for (int p = 0; p < LSEQ; p++) s += __expf(base[(long long)p * LSEQ] - m);
  cmax[(long long)b * LSEQ + q] = m;
  csum[(long long)b * LSEQ + q] = s;
}

// Write aligned_p bf16 [p][q] (row softmax) and aligned_q bf16 [q][p]
// (col softmax, transposed via LDS) from one 64x64 tile read of affinity.
__global__ __launch_bounds__(256) void softmax_write(
    const float* __restrict__ aff, const float* __restrict__ rmax,
    const float* __restrict__ rsum, const float* __restrict__ cmax,
    const float* __restrict__ csum, unsigned short* __restrict__ alP,
    unsigned short* __restrict__ alQ) {
  __shared__ unsigned short tq[64][65];
  const int b = blockIdx.z;
  const int p0 = blockIdx.y * 64, q0 = blockIdx.x * 64;
  const int tx = threadIdx.x & 63, ty = threadIdx.x >> 6;
  const float* A = aff + (long long)b * LSEQ * LSEQ;
  const long long bl = (long long)b * LSEQ;
  const int q = q0 + tx;
  const float cm = cmax[bl + q];
  const float ci = 1.f / csum[bl + q];
#pragma unroll
  for (int i = 0; i < 16; i++) {
    int pr = ty + i * 4;
    int p = p0 + pr;
    float a = A[(long long)p * LSEQ + q];
    float rm = rmax[bl + p];
    float ri = 1.f / rsum[bl + p];
    alP[(bl + p) * (long long)LSEQ + q] = f2bf(__expf(a - rm) * ri);
    tq[tx][pr] = f2bf(__expf(a - cm) * ci);  // tq[q_local][p_local]
  }
  __syncthreads();
#pragma unroll
  for (int i = 0; i < 16; i++) {
    int qr = ty + i * 4;
    alQ[(bl + q0 + qr) * (long long)LSEQ + p0 + tx] = tq[qr][tx];
  }
}

// Six FM heads. blockIdx.y = side (0: passage features c1,c3,c5 using
// aligned=query_aligned & base=passage; 1: query features c2,c4,c6).
// One wave32 per sequence position; lanes stride over D, then shfl-reduce.
__global__ __launch_bounds__(256) void fm_features(
    const float* __restrict__ stack, const float* __restrict__ pAligned,
    const float* __restrict__ qAligned, const float* __restrict__ cat_w0,
    const float* __restrict__ cat_w, const float* __restrict__ cat_V,
    const float* __restrict__ dm_w0, const float* __restrict__ dm_w,
    const float* __restrict__ dm_V, float* __restrict__ out) {
  const int wave = threadIdx.x >> 5, lane = threadIdx.x & 31;
  const long long pos = (long long)blockIdx.x * 8 + wave;
  const int s = blockIdx.y;
  const float* alg = (s == 0 ? qAligned : pAligned) + pos * D_PAD;
  const float* bse = stack + (long long)s * BL * D_IN + pos * D_IN;
  const float* wc = cat_w + s * 2 * D_IN;
  const float* Vc = cat_V + (long long)s * KFM * 2 * D_IN;
  const float* wd = dm_w + s * D_IN;
  const float* Vd = dm_V + (long long)s * KFM * D_IN;
  const float* wmu = dm_w + (2 + s) * D_IN;
  const float* Vm = dm_V + (long long)(2 + s) * KFM * D_IN;

  float linc = 0.f, lind = 0.f, linm = 0.f;
  float d1c[KFM], d2c[KFM], d1d[KFM], d2d[KFM], d1m[KFM], d2m[KFM];
#pragma unroll
  for (int k = 0; k < KFM; k++) {
    d1c[k] = d2c[k] = d1d[k] = d2d[k] = d1m[k] = d2m[k] = 0.f;
  }
  for (int d = lane; d < D_IN; d += 32) {
    float a = alg[d], bs = bse[d];
    float xd = a - bs, xm = a * bs;
    linc += a * wc[d] + bs * wc[D_IN + d];
    lind += xd * wd[d];
    linm += xm * wmu[d];
    float a2 = a * a, b2 = bs * bs, xd2 = xd * xd, xm2 = xm * xm;
#pragma unroll
    for (int k = 0; k < KFM; k++) {
      float v1 = Vc[k * 2 * D_IN + d];
      float v2 = Vc[k * 2 * D_IN + D_IN + d];
      d1c[k] += a * v1 + bs * v2;
      d2c[k] += a2 * v1 * v1 + b2 * v2 * v2;
      float vd = Vd[k * D_IN + d];
      d1d[k] += xd * vd; d2d[k] += xd2 * vd * vd;
      float vm = Vm[k * D_IN + d];
      d1m[k] += xm * vm; d2m[k] += xm2 * vm * vm;
    }
  }
  for (int o = 16; o; o >>= 1) {
    linc += __shfl_xor(linc, o, 32);
    lind += __shfl_xor(lind, o, 32);
    linm += __shfl_xor(linm, o, 32);
#pragma unroll
    for (int k = 0; k < KFM; k++) {
      d1c[k] += __shfl_xor(d1c[k], o, 32);
      d2c[k] += __shfl_xor(d2c[k], o, 32);
      d1d[k] += __shfl_xor(d1d[k], o, 32);
      d2d[k] += __shfl_xor(d2d[k], o, 32);
      d1m[k] += __shfl_xor(d1m[k], o, 32);
      d2m[k] += __shfl_xor(d2m[k], o, 32);
    }
  }
  if (lane == 0) {
    float ic = 0.f, idd = 0.f, im = 0.f;
#pragma unroll
    for (int k = 0; k < KFM; k++) {
      ic += d1c[k] * d1c[k] - d2c[k];
      idd += d1d[k] * d1d[k] - d2d[k];
      im += d1m[k] * d1m[k] - d2m[k];
    }
    float cc = cat_w0[s] + linc + 0.5f * ic;
    float cd = dm_w0[s] + lind + 0.5f * idd;
    float cm = dm_w0[2 + s] + linm + 0.5f * im;
    float* o = out + (long long)s * BL * 3 + pos * 3;
    o[0] = cc; o[1] = cd; o[2] = cm;
  }
}

extern "C" void kernel_launch(void* const* d_in, const int* in_sizes, int n_in,
                              void* d_out, int out_size, void* d_ws,
                              size_t ws_size, hipStream_t stream) {
  const float* stack  = (const float*)d_in[0];
  const float* W1     = (const float*)d_in[1];
  const float* b1     = (const float*)d_in[2];
  const float* W2     = (const float*)d_in[3];
  const float* b2     = (const float*)d_in[4];
  const float* catw0  = (const float*)d_in[5];
  const float* catw   = (const float*)d_in[6];
  const float* catV   = (const float*)d_in[7];
  const float* dmw0   = (const float*)d_in[8];
  const float* dmw    = (const float*)d_in[9];
  const float* dmV    = (const float*)d_in[10];

  char* ws = (char*)d_ws;
  size_t off = 0;
  auto alloc = [&](size_t bytes) -> char* {
    char* p = ws + off;
    off = (off + bytes + 255) & ~(size_t)255;
    return p;
  };

  unsigned short* Xbf  = (unsigned short*)alloc(2ll * BL * D_PAD * 2);
  unsigned short* XTbf = (unsigned short*)alloc(2ll * BATCH * D_PAD * LSEQ * 2);
  unsigned short* Wt   = (unsigned short*)alloc(2ll * U_PAD * D_PAD * 2);
  float* biasP         = (float*)alloc(2ll * U_PAD * 4);
  unsigned short* denP = (unsigned short*)alloc((long long)BL * U_PAD * 2);
  unsigned short* denQ = (unsigned short*)alloc((long long)BL * U_PAD * 2);
  float* aff           = (float*)alloc((long long)BATCH * LSEQ * LSEQ * 4);
  float* rmax          = (float*)alloc((long long)BL * 4);
  float* rsum          = (float*)alloc((long long)BL * 4);
  float* cmax          = (float*)alloc((long long)BL * 4);
  float* csum          = (float*)alloc((long long)BL * 4);
  unsigned short* alP  = (unsigned short*)alloc((long long)BATCH * LSEQ * LSEQ * 2);
  unsigned short* alQ  = (unsigned short*)alloc((long long)BATCH * LSEQ * LSEQ * 2);
  float* pAligned      = (float*)alloc((long long)BL * D_PAD * 4);
  float* qAligned      = (float*)alloc((long long)BL * D_PAD * 4);
  (void)ws_size; (void)in_sizes; (void)n_in; (void)out_size;

  // 1) param conversion
  convert_weights<<<dim3((2 * U_PAD * D_PAD + 255) / 256), 256, 0, stream>>>(
      W1, W2, Wt);
  convert_bias<<<dim3(3), 256, 0, stream>>>(b1, b2, biasP);
  // 2) input conversion (+transpose)
  convert_inputs<<<dim3(LSEQ / 64, D_PAD / 64, 2 * BATCH), 256, 0, stream>>>(
      stack, Xbf, XTbf);
  // 3) dense projections, relu, bf16 out
  gemm_wmma<0><<<dim3(U_PAD / 64, LSEQ / 128, BATCH), GEMM_THREADS, 0, stream>>>(
      Xbf, Wt, denP, biasP, 1.f, D_PAD, D_PAD, D_PAD, U_PAD,
      (long long)LSEQ * D_PAD, 0ll, (long long)LSEQ * U_PAD);
  gemm_wmma<0><<<dim3(U_PAD / 64, LSEQ / 128, BATCH), GEMM_THREADS, 0, stream>>>(
      Xbf + (long long)BL * D_PAD, Wt + U_PAD * D_PAD, denQ, biasP + U_PAD, 1.f,
      D_PAD, D_PAD, D_PAD, U_PAD, (long long)LSEQ * D_PAD, 0ll,
      (long long)LSEQ * U_PAD);
  // 4) affinity = scale * p_dense @ q_dense^T (f32, lives in the 192MB L2)
  const float scale = 1.f / sqrtf((float)D_IN);
  gemm_wmma<1><<<dim3(LSEQ / 64, LSEQ / 128, BATCH), GEMM_THREADS, 0, stream>>>(
      denP, denQ, aff, nullptr, scale, U_PAD, U_PAD, U_PAD, LSEQ,
      (long long)LSEQ * U_PAD, (long long)LSEQ * U_PAD,
      (long long)LSEQ * LSEQ);
  // 5) softmax stats (row and column)
  row_stats<<<dim3(LSEQ, BATCH), 256, 0, stream>>>(aff, rmax, rsum);
  col_stats<<<dim3(LSEQ / 256, BATCH), 256, 0, stream>>>(aff, cmax, csum);
  // 6) write aligned_p [p][q] and aligned_q [q][p] in bf16
  softmax_write<<<dim3(LSEQ / 64, LSEQ / 64, BATCH), 256, 0, stream>>>(
      aff, rmax, rsum, cmax, csum, alP, alQ);
  // 7) passage_aligned = aligned_p @ passage ; query_aligned = aligned_q @ query
  gemm_wmma<2><<<dim3(D_PAD / 64, LSEQ / 128, BATCH), GEMM_THREADS, 0, stream>>>(
      alP, XTbf, pAligned, nullptr, 1.f, LSEQ, LSEQ, LSEQ, D_PAD,
      (long long)LSEQ * LSEQ, (long long)D_PAD * LSEQ,
      (long long)LSEQ * D_PAD);
  gemm_wmma<2><<<dim3(D_PAD / 64, LSEQ / 128, BATCH), GEMM_THREADS, 0, stream>>>(
      alQ, XTbf + (long long)BATCH * D_PAD * LSEQ, qAligned, nullptr, 1.f,
      LSEQ, LSEQ, LSEQ, D_PAD, (long long)LSEQ * LSEQ,
      (long long)D_PAD * LSEQ, (long long)LSEQ * D_PAD);
  // 8) FM heads -> d_out (features_passage then features_query)
  fm_features<<<dim3(BL / 8, 2), 256, 0, stream>>>(
      stack, pAligned, qAligned, catw0, catw, catV, dmw0, dmw, dmV,
      (float*)d_out);
}
